// GPT_53858889892364
// MI455X (gfx1250) — compile-verified, matrix operand order checked by reading, not run on hardware
//
#include <hip/hip_runtime.h>
#include <math.h>

// ---------------- types ----------------
typedef __bf16 bf16_t;
typedef bf16_t v16bf  __attribute__((ext_vector_type(16)));
typedef bf16_t bf16x8 __attribute__((ext_vector_type(8)));
typedef float  v8f    __attribute__((ext_vector_type(8)));

// ---------------- model constants ----------------
constexpr int NL = 12;        // layers
constexpr int HH = 12;        // heads
constexpr int Ee = 768;       // embed
constexpr int Tt = 1024;      // seq len
constexpr int BB = 2;         // batch
constexpr int VV = 50257;     // vocab
constexpr int HS = 64;        // head size
constexpr int E3 = 3 * Ee;    // 2304
constexpr int E4 = 4 * Ee;    // 3072
constexpr int MM = BB * Tt;   // 2048 tokens
constexpr float LN_EPS = 1e-5f;

// Async global->LDS copy of 16 bytes (CDNA5 GLOBAL_LOAD_ASYNC_TO_LDS_B128,
// tracked by ASYNCcnt). ldsaddr = LDS byte offset (low 32 bits of generic
// shared pointer), gaddr = 64-bit global address.
__device__ __forceinline__ void async_copy_b128(unsigned ldsaddr, const void* gaddr) {
  asm volatile("global_load_async_to_lds_b128 %0, %1, off"
               :: "v"(ldsaddr), "v"(gaddr) : "memory");
}
__device__ __forceinline__ void wait_async0() {
  asm volatile("s_wait_asynccnt 0x0" ::: "memory");
}

// ======================================================================
// Elementwise / conversion kernels
// ======================================================================
__global__ void embed_kernel(const int* __restrict__ idx,
                             const float* __restrict__ wte,
                             const float* __restrict__ wpe,
                             float* __restrict__ x, int total) {
  int i = blockIdx.x * blockDim.x + threadIdx.x;
  if (i >= total) return;
  int e  = i % Ee;
  int bt = i / Ee;
  int t  = bt % Tt;
  x[i] = wte[(long)idx[bt] * Ee + e] + wpe[(long)t * Ee + e];
}

__global__ void cvt_f32_bf16_kernel(const float* __restrict__ in,
                                    bf16_t* __restrict__ out, long total) {
  long i = (long)blockIdx.x * blockDim.x + threadIdx.x;
  if (i < total) out[i] = (bf16_t)in[i];
}

// out[l][n][k] = in[l][k][n]   (in: [L][K][N] fp32, out: [L][N][K] bf16)
__global__ void cvt_transpose_kernel(const float* __restrict__ in,
                                     bf16_t* __restrict__ out,
                                     int K, int N, long total) {
  long i = (long)blockIdx.x * blockDim.x + threadIdx.x;
  if (i >= total) return;
  long per = (long)K * N;
  long l = i / per;
  long j = i - l * per;
  int n = (int)(j / K);
  int k = (int)(j - (long)n * K);
  out[i] = (bf16_t)in[l * per + (long)k * N + n];
}

// ======================================================================
// LayerNorm: fp32 residual in -> bf16 normalized out (one block per row)
// ======================================================================
__global__ __launch_bounds__(256) void layernorm_bf16_kernel(
    const float* __restrict__ x, const float* __restrict__ w,
    const float* __restrict__ b, bf16_t* __restrict__ out) {
  __shared__ float red[256];
  const int row = blockIdx.x, tid = threadIdx.x;
  const float* xr = x + (long)row * Ee;

  float s = 0.f;
  for (int i = tid; i < Ee; i += 256) s += xr[i];
  red[tid] = s; __syncthreads();
  for (int st = 128; st > 0; st >>= 1) {
    if (tid < st) red[tid] += red[tid + st];
    __syncthreads();
  }
  const float mean = red[0] * (1.0f / Ee);
  __syncthreads();

  float v = 0.f;
  for (int i = tid; i < Ee; i += 256) { float d = xr[i] - mean; v += d * d; }
  red[tid] = v; __syncthreads();
  for (int st = 128; st > 0; st >>= 1) {
    if (tid < st) red[tid] += red[tid + st];
    __syncthreads();
  }
  const float rs = rsqrtf(red[0] * (1.0f / Ee) + LN_EPS);

  bf16_t* orow = out + (long)row * Ee;
  for (int i = tid; i < Ee; i += 256)
    orow[i] = (bf16_t)((xr[i] - mean) * rs * w[i] + b[i]);
}

// ======================================================================
// Causal softmax with 1/sqrt(HS) scale: fp32 scores -> bf16 probabilities
// One block per (b,h,q) row of length T.
// ======================================================================
__global__ __launch_bounds__(256) void softmax_causal_kernel(
    const float* __restrict__ S, bf16_t* __restrict__ P) {
  __shared__ float red[256];
  const int row = blockIdx.x;          // (b*H + h)*T + q
  const int q   = row % Tt;
  const int tid = threadIdx.x;
  const float scale = 0.125f;          // 1/sqrt(64)
  const float* sr = S + (long)row * Tt;

  float m = -3.4e38f;
  for (int k = tid; k <= q; k += 256) m = fmaxf(m, sr[k] * scale);
  red[tid] = m; __syncthreads();
  for (int st = 128; st > 0; st >>= 1) {
    if (tid < st) red[tid] = fmaxf(red[tid], red[tid + st]);
    __syncthreads();
  }
  m = red[0]; __syncthreads();

  float sum = 0.f;
  for (int k = tid; k <= q; k += 256) sum += __expf(sr[k] * scale - m);
  red[tid] = sum; __syncthreads();
  for (int st = 128; st > 0; st >>= 1) {
    if (tid < st) red[tid] += red[tid + st];
    __syncthreads();
  }
  const float inv = 1.f / red[0];

  bf16_t* pr = P + (long)row * Tt;
  for (int k = tid; k < Tt; k += 256) {
    float pv = (k <= q) ? __expf(sr[k] * scale - m) * inv : 0.f;
    pr[k] = (bf16_t)pv;
  }
}

// ======================================================================
// Tiled WMMA GEMM with double-buffered async global->LDS staging.
//   C[m,n] (+)= sum_k A[m,k]*B[k,n]  (+ bias, gelu, bf16 output)
//   A: bf16 [M,K] row-major (lda), M multiple of 128, K multiple of 32
//   B_NK=true : B supplied as BT[N,K] row-major (async staging, N-guarded)
//   B_NK=false: B supplied as B[K,N] row-major (sync transposed staging)
//   C: fp32 or bf16 (OUT_BF16); strides cOut/cIn/ldc are in C elements.
//   Batched via grid.z with decomposition z = zo*innerN + zi.
// Block: 128 threads = 4 waves (2x2); tile 128x128; wave tile 64x64
// (4x4 WMMA accumulators = 128 VGPRs; 16 WMMAs per 16 ds_load_b128).
// ======================================================================
template <bool B_NK, bool HAS_BIAS, bool ACCUM, bool DO_GELU, bool OUT_BF16>
__global__ __launch_bounds__(128, 1) void gemm_bf16_wmma(
    const bf16_t* __restrict__ A, long aOut, long aIn,
    const bf16_t* __restrict__ B, long bOut, long bIn,
    void* __restrict__ C, long cOut, long cIn,
    const float* __restrict__ bias,
    int N, int K, int lda, int ldb, int ldc, int innerN) {
  // LDS: row stride 48 bf16 (96B) => each 16-element fragment slice 32B-aligned
  __shared__ bf16_t As[2][128 * 48];
  __shared__ bf16_t Bs[2][128 * 48];

  const int z  = blockIdx.z;
  const int zo = z / innerN;
  const int zi = z - zo * innerN;
  const bf16_t* Ab  = A + zo * aOut + zi * aIn;
  const bf16_t* Bbp = B + zo * bOut + zi * bIn;

  const int tid  = threadIdx.x;
  const int wave = tid >> 5;
  const int lane = tid & 31;
  const int wm   = (wave >> 1) * 64;   // wave M offset in tile
  const int wn   = (wave & 1) * 64;    // wave N offset in tile
  const int m0   = blockIdx.y * 128;
  const int n0   = blockIdx.x * 128;
  const int lr   = lane & 15;          // row-in-fragment
  const int lh   = lane >> 4;          // K-half / M-half selector
  const bool fullN = (n0 + 128) <= N;  // uniform: interior tile in N

  // stage A tile (async): 128 rows x 32 cols, 16B chunks, 4 per thread
  auto stageA = [&](int k0, int buf) {
#pragma unroll
    for (int it = 0; it < 4; it++) {
      int i = tid + it * 128;
      int r = i >> 2, c = (i & 3) * 8;
      unsigned lds = (unsigned)(unsigned long)&As[buf][r * 48 + c];
      async_copy_b128(lds, Ab + (long)(m0 + r) * lda + (k0 + c));
    }
  };
  // stage B tile into Bs[buf][n][k]
  auto stageB = [&](int k0, int buf) {
    if (B_NK) {
#pragma unroll
      for (int it = 0; it < 4; it++) {
        int i = tid + it * 128;
        int r = i >> 2, c = (i & 3) * 8;
        if (fullN || (n0 + r < N)) {
          unsigned lds = (unsigned)(unsigned long)&Bs[buf][r * 48 + c];
          async_copy_b128(lds, Bbp + (long)(n0 + r) * ldb + (k0 + c));
        } else {
          bf16x8 zv;
#pragma unroll
          for (int j = 0; j < 8; j++) zv[j] = (bf16_t)0.0f;
          *(bf16x8*)(&Bs[buf][r * 48 + c]) = zv;
        }
      }
    } else {
#pragma unroll
      for (int it = 0; it < 4; it++) {
        int i  = tid + it * 128;
        int kk = i >> 4;           // 0..31
        int nn = (i & 15) * 8;     // 0..120
#pragma unroll
        for (int j = 0; j < 8; j++) {
          int n = n0 + nn + j;
          bf16_t v = (bf16_t)0.0f;
          if (n < N) v = Bbp[(long)(k0 + kk) * ldb + n];
          Bs[buf][(nn + j) * 48 + kk] = v;
        }
      }
    }
  };

  v8f acc[4][4];
#pragma unroll
  for (int i = 0; i < 4; i++)
#pragma unroll
    for (int j = 0; j < 4; j++)
#pragma unroll
      for (int r = 0; r < 8; r++) acc[i][j][r] = 0.f;

  // prologue: fill buffer 0
  stageA(0, 0);
  stageB(0, 0);
  wait_async0();
  __syncthreads();

  int cur = 0;
  for (int k0 = 0; k0 < K; k0 += 32) {
    const int nxt = cur ^ 1;
    if (k0 + 32 < K) {           // prefetch next tile while computing
      stageA(k0 + 32, nxt);
      stageB(k0 + 32, nxt);
    }

    // ---- load fragments from current buffer and issue WMMAs ----
    v16bf af[4], bfv[4];
#pragma unroll
    for (int mf = 0; mf < 4; mf++)
      af[mf] = *(const v16bf*)(&As[cur][(wm + mf * 16 + lr) * 48 + lh * 16]);
#pragma unroll
    for (int nf = 0; nf < 4; nf++)
      bfv[nf] = *(const v16bf*)(&Bs[cur][(wn + nf * 16 + lr) * 48 + lh * 16]);
#pragma unroll
    for (int mf = 0; mf < 4; mf++)
#pragma unroll
      for (int nf = 0; nf < 4; nf++)
        acc[mf][nf] = __builtin_amdgcn_wmma_f32_16x16x32_bf16(
            false, af[mf], false, bfv[nf], (short)0, acc[mf][nf], false, false);

    wait_async0();               // next buffer's async copies landed in LDS
    __syncthreads();             // visible to all waves; cur free for reuse
    cur = nxt;
  }

  // ---- epilogue: bias / gelu / residual-accumulate / store ----
#pragma unroll
  for (int mf = 0; mf < 4; mf++) {
#pragma unroll
    for (int nf = 0; nf < 4; nf++) {
      int n = n0 + wn + nf * 16 + lr;
      if (fullN || n < N) {
        float bv = HAS_BIAS ? bias[n] : 0.f;
#pragma unroll
        for (int r = 0; r < 8; r++) {
          int m = m0 + wm + mf * 16 + lh * 8 + r;
          float v = acc[mf][nf][r] + bv;
          if (DO_GELU) v = 0.5f * v * (1.f + erff(v * 0.70710678118f));
          if (OUT_BF16) {
            bf16_t* cp = (bf16_t*)C + zo * cOut + zi * cIn + (long)m * ldc + n;
            *cp = (bf16_t)v;
          } else {
            float* cp = (float*)C + zo * cOut + zi * cIn + (long)m * ldc + n;
            if (ACCUM) *cp = *cp + v; else *cp = v;
          }
        }
      }
    }
  }
}

// ======================================================================
// Host orchestration
// ======================================================================
extern "C" void kernel_launch(void* const* d_in, const int* in_sizes, int n_in,
                              void* d_out, int out_size, void* d_ws, size_t ws_size,
                              hipStream_t stream) {
  (void)in_sizes; (void)n_in; (void)out_size; (void)ws_size;

  const int*   idx    = (const int*)  d_in[0];
  const float* wte    = (const float*)d_in[1];
  const float* wpe    = (const float*)d_in[2];
  const float* ln1_w  = (const float*)d_in[3];
  const float* ln1_b  = (const float*)d_in[4];
  const float* attn_w = (const float*)d_in[5];
  const float* attn_b = (const float*)d_in[6];
  const float* proj_w = (const float*)d_in[7];
  const float* proj_b = (const float*)d_in[8];
  const float* ln2_w  = (const float*)d_in[9];
  const float* ln2_b  = (const float*)d_in[10];
  const float* fc_w   = (const float*)d_in[11];
  const float* fc_b   = (const float*)d_in[12];
  const float* fcp_w  = (const float*)d_in[13];
  const float* fcp_b  = (const float*)d_in[14];
  const float* lnf_w  = (const float*)d_in[15];
  const float* lnf_b  = (const float*)d_in[16];
  float* out = (float*)d_out;

  // ---- workspace bump allocator (256B aligned) ----
  char* wp = (char*)d_ws;
  auto alloc = [&](size_t bytes) -> void* {
    void* r = (void*)wp;
    wp += (bytes + 255) & ~(size_t)255;
    return r;
  };
  bf16_t* wteb = (bf16_t*)alloc((size_t)VV * Ee * 2);        // [V,E]  (== BT for lm_head)
  bf16_t* awT  = (bf16_t*)alloc((size_t)NL * E3 * Ee * 2);   // [L][3E][E]
  bf16_t* pwT  = (bf16_t*)alloc((size_t)NL * Ee * Ee * 2);   // [L][E][E]
  bf16_t* fwT  = (bf16_t*)alloc((size_t)NL * E4 * Ee * 2);   // [L][4E][E]
  bf16_t* fpwT = (bf16_t*)alloc((size_t)NL * Ee * E4 * 2);   // [L][E][4E]
  float*  x    = (float*) alloc((size_t)MM * Ee * 4);        // fp32 residual stream
  bf16_t* xn   = (bf16_t*)alloc((size_t)MM * Ee * 2);        // LN output (bf16)
  bf16_t* qkvb = (bf16_t*)alloc((size_t)MM * E3 * 2);        // qkv (bf16, GEMM direct)
  float*  Sbuf = (float*) alloc((size_t)BB * HH * Tt * Tt * 4);
  bf16_t* Pbuf = (bf16_t*)alloc((size_t)BB * HH * Tt * Tt * 2);
  bf16_t* yb   = (bf16_t*)alloc((size_t)MM * Ee * 2);        // attn out (bf16)
  bf16_t* hbb  = (bf16_t*)alloc((size_t)MM * E4 * 2);        // gelu(fc) (bf16)

  // ---- weight precision conversion / transposition (per call) ----
  { long t = (long)VV * Ee;
    cvt_f32_bf16_kernel<<<(t + 255) / 256, 256, 0, stream>>>(wte, wteb, t); }
  { long t = (long)NL * E3 * Ee;
    cvt_transpose_kernel<<<(t + 255) / 256, 256, 0, stream>>>(attn_w, awT, Ee, E3, t); }
  { long t = (long)NL * Ee * Ee;
    cvt_transpose_kernel<<<(t + 255) / 256, 256, 0, stream>>>(proj_w, pwT, Ee, Ee, t); }
  { long t = (long)NL * E4 * Ee;
    cvt_transpose_kernel<<<(t + 255) / 256, 256, 0, stream>>>(fc_w, fwT, Ee, E4, t); }
  { long t = (long)NL * Ee * E4;
    cvt_transpose_kernel<<<(t + 255) / 256, 256, 0, stream>>>(fcp_w, fpwT, E4, Ee, t); }

  // ---- embedding ----
  embed_kernel<<<(MM * Ee + 255) / 256, 256, 0, stream>>>(idx, wte, wpe, x, MM * Ee);

  const long zA_out = (long)Tt * E3;   // per-batch stride inside qkv
  const long zA_in  = (long)HS;        // per-head stride inside qkv

  for (int l = 0; l < NL; l++) {
    // LN1 -> xn (bf16)
    layernorm_bf16_kernel<<<MM, 256, 0, stream>>>(x, ln1_w + (long)l * Ee,
                                                  ln1_b + (long)l * Ee, xn);
    // qkvb = bf16(xn @ attn_w + attn_b)   (direct bf16 output)
    gemm_bf16_wmma<true, true, false, false, true>
        <<<dim3(E3 / 128, MM / 128, 1), 128, 0, stream>>>(
            xn, 0L, 0L, awT + (long)l * E3 * Ee, 0L, 0L, qkvb, 0L, 0L,
            attn_b + (long)l * E3, E3, Ee, Ee, Ee, E3, 1);

    // S = Q @ K^T   (batched over B*H=24; K matrix is already [N=T, K=HS])
    gemm_bf16_wmma<true, false, false, false, false>
        <<<dim3(Tt / 128, Tt / 128, BB * HH), 128, 0, stream>>>(
            qkvb,        zA_out, zA_in,
            qkvb + Ee,   zA_out, zA_in,
            Sbuf, (long)HH * Tt * Tt, (long)Tt * Tt,
            nullptr, Tt, HS, E3, E3, Tt, HH);

    // causal softmax (with 1/sqrt(HS) scale) -> P (bf16)
    softmax_causal_kernel<<<BB * HH * Tt, 256, 0, stream>>>(Sbuf, Pbuf);

    // yb = bf16(P @ V)   (B supplied as [K=T, N=HS], N=64 edge-guarded)
    gemm_bf16_wmma<false, false, false, false, true>
        <<<dim3(1, Tt / 128, BB * HH), 128, 0, stream>>>(
            Pbuf, (long)HH * Tt * Tt, (long)Tt * Tt,
            qkvb + 2 * Ee, zA_out, zA_in,
            yb, (long)Tt * Ee, (long)HS,
            nullptr, HS, Tt, Tt, E3, Ee, HH);

    // x += yb @ proj_w + proj_b   (fp32 residual accumulate)
    gemm_bf16_wmma<true, true, true, false, false>
        <<<dim3(Ee / 128, MM / 128, 1), 128, 0, stream>>>(
            yb, 0L, 0L, pwT + (long)l * Ee * Ee, 0L, 0L, x, 0L, 0L,
            proj_b + (long)l * Ee, Ee, Ee, Ee, Ee, Ee, 1);

    // LN2 -> xn
    layernorm_bf16_kernel<<<MM, 256, 0, stream>>>(x, ln2_w + (long)l * Ee,
                                                  ln2_b + (long)l * Ee, xn);
    // hbb = bf16(gelu(xn @ fc_w + fc_b))   (direct bf16 output)
    gemm_bf16_wmma<true, true, false, true, true>
        <<<dim3(E4 / 128, MM / 128, 1), 128, 0, stream>>>(
            xn, 0L, 0L, fwT + (long)l * E4 * Ee, 0L, 0L, hbb, 0L, 0L,
            fc_b + (long)l * E4, E4, Ee, Ee, Ee, E4, 1);

    // x += hbb @ fcp_w + fcp_b
    gemm_bf16_wmma<true, true, true, false, false>
        <<<dim3(Ee / 128, MM / 128, 1), 128, 0, stream>>>(
            hbb, 0L, 0L, fpwT + (long)l * Ee * E4, 0L, 0L, x, 0L, 0L,
            fcp_b + (long)l * Ee, Ee, E4, E4, E4, Ee, 1);
  }

  // final LN -> xn
  layernorm_bf16_kernel<<<MM, 256, 0, stream>>>(x, lnf_w, lnf_b, xn);

  // logits = xn @ wte^T  (wteb [V,E] is exactly the BT layout; N edge-guarded)
  gemm_bf16_wmma<true, false, false, false, false>
      <<<dim3((VV + 127) / 128, MM / 128, 1), 128, 0, stream>>>(
          xn, 0L, 0L, wteb, 0L, 0L, out, 0L, 0L,
          nullptr, VV, Ee, Ee, Ee, VV, 1);
}
